// BaselineModel_5540507811855
// MI455X (gfx1250) — compile-verified, matrix operand order checked by reading, not run on hardware
//
#include <hip/hip_runtime.h>

// ---------------------------------------------------------------------------
// Types for WMMA fragments (gfx1250, wave32)
// ---------------------------------------------------------------------------
typedef __attribute__((ext_vector_type(4)))  unsigned int u32x4;
typedef __attribute__((ext_vector_type(8)))  unsigned int u32x8;
typedef __attribute__((ext_vector_type(16))) __bf16      v16bf;
typedef __attribute__((ext_vector_type(8)))  float       v8f;

__device__ __forceinline__ unsigned short f2bf(float f) {
  unsigned u = __float_as_uint(f);
  u += 0x7FFFu + ((u >> 16) & 1u);          // round-to-nearest-even
  return (unsigned short)(u >> 16);
}
__device__ __forceinline__ float bf2f(unsigned short h) {
  return __uint_as_float(((unsigned)h) << 16);
}

// Load one 16x32-bf16 fragment half-pair per ISA layout:
// lanes 0-15 take K[0..7]/K[16..23], lanes 16-31 take K[8..15]/K[24..31].
__device__ __forceinline__ u32x8 ldfrag(const unsigned short* p, int ko1) {
  u32x4 lo = *(const u32x4*)(p + ko1);
  u32x4 hi = *(const u32x4*)(p + ko1 + 16);
  return __builtin_shufflevector(lo, hi, 0, 1, 2, 3, 4, 5, 6, 7);
}
__device__ __forceinline__ u32x8 ldfrag_cond(const unsigned short* p, int ko1,
                                             bool v) {
  u32x8 z = {0u, 0u, 0u, 0u, 0u, 0u, 0u, 0u};
  return v ? ldfrag(p, ko1) : z;
}
__device__ __forceinline__ v8f wmma_bf16(u32x8 a, u32x8 b, v8f c) {
  return __builtin_amdgcn_wmma_f32_16x16x32_bf16(
      false, __builtin_bit_cast(v16bf, a), false, __builtin_bit_cast(v16bf, b),
      (short)0, c, false, false);
}

// ---------------------------------------------------------------------------
// Packing kernels (fp32 -> bf16 bit buffers)
// ---------------------------------------------------------------------------
__global__ void pack_bf16(unsigned short* __restrict__ dst,
                          const float* __restrict__ src, int total) {
  int tid = blockIdx.x * blockDim.x + threadIdx.x;
  if (tid < total) dst[tid] = f2bf(src[tid]);
}

// conv1 weights: [256][1][3][3] -> [co][k] with K padded 9 -> 32
__global__ void pack_w1(unsigned short* __restrict__ dst,
                        const float* __restrict__ w) {
  int tid = blockIdx.x * blockDim.x + threadIdx.x;   // 256*32
  int k = tid & 31, co = tid >> 5;
  dst[tid] = f2bf(k < 9 ? w[co * 9 + k] : 0.f);
}

// conv weights OIHW [Cout][Cin][3][3] -> kn2row layout [rc][Cout][Cin]
__global__ void pack_conv_w(unsigned short* __restrict__ dst,
                            const float* __restrict__ w,
                            int Cout, int Cin, int total) {
  int tid = blockIdx.x * blockDim.x + threadIdx.x;
  if (tid >= total) return;
  int rc = tid % 9;
  int t  = tid / 9;
  int ci = t % Cin;
  int co = t / Cin;
  dst[((size_t)rc * Cout + co) * Cin + ci] = f2bf(w[tid]);
}

// conv1 im2col: x [128][1][48][48] f32 -> col [294912][32] bf16 (K padded)
__global__ void im2col_conv1(unsigned short* __restrict__ col,
                             const float* __restrict__ x) {
  int tid = blockIdx.x * blockDim.x + threadIdx.x;   // 294912*32
  int k = tid & 31, p = tid >> 5;
  float v = 0.f;
  if (k < 9) {
    int b = p / 2304;
    int rem = p - b * 2304;
    int h = rem / 48, w = rem - (rem / 48) * 48;
    int ih = h + k / 3 - 1, iw = w + (k % 3) - 1;
    if ((unsigned)ih < 48u && (unsigned)iw < 48u)
      v = x[b * 2304 + ih * 48 + iw];
  }
  col[tid] = f2bf(v);
}

// ---------------------------------------------------------------------------
// Generic bf16 GEMM with WMMA:  y[M][N] = A[M][K] * Bt[N][K]^T + bias
// block = 128 threads (4 waves); wave -> 16x64 tile. M%64==0, N%64==0, K%32==0
// Software-pipelined: fragments for step k+32 are issued before the WMMAs
// consuming step k, so the wave overlaps VMEM with the matrix pipe.
// ---------------------------------------------------------------------------
__global__ __launch_bounds__(128) void gemm_wmma(
    float* __restrict__ y, const unsigned short* __restrict__ A,
    const unsigned short* __restrict__ Bt, const float* __restrict__ bias,
    int N, int K) {
  const int lane  = threadIdx.x & 31;
  const int wave  = threadIdx.x >> 5;
  const int m0    = (blockIdx.y * 4 + wave) * 16;
  const int n0    = blockIdx.x * 64;
  const int lm    = lane & 15;
  const int hiSel = lane >> 4;
  const int ko1   = hiSel * 8;          // ISA 16-bit A/B fragment K offset

  const unsigned short* arow = A + (size_t)(m0 + lm) * K;
  const unsigned short* brow[4];
#pragma unroll
  for (int t = 0; t < 4; ++t)
    brow[t] = Bt + (size_t)(n0 + t * 16 + lm) * K;

  v8f z = {0.f, 0.f, 0.f, 0.f, 0.f, 0.f, 0.f, 0.f};
  v8f acc[4] = {z, z, z, z};

  // prologue: fragments for k = 0
  u32x8 aC = ldfrag(arow, ko1);
  u32x8 bC[4];
#pragma unroll
  for (int t = 0; t < 4; ++t) bC[t] = ldfrag(brow[t], ko1);

  for (int k = 32; k < K; k += 32) {
    u32x8 aN = ldfrag(arow + k, ko1);
    u32x8 bN[4];
#pragma unroll
    for (int t = 0; t < 4; ++t) bN[t] = ldfrag(brow[t] + k, ko1);
#pragma unroll
    for (int t = 0; t < 4; ++t) acc[t] = wmma_bf16(aC, bC[t], acc[t]);
    aC = aN;
#pragma unroll
    for (int t = 0; t < 4; ++t) bC[t] = bN[t];
  }
#pragma unroll
  for (int t = 0; t < 4; ++t) acc[t] = wmma_bf16(aC, bC[t], acc[t]);

  const int rbase = m0 + hiSel * 8;
#pragma unroll
  for (int t = 0; t < 4; ++t) {
    const int col = n0 + t * 16 + lm;
#pragma unroll
    for (int i = 0; i < 8; ++i) {
      const int row = rbase + i;
      y[(size_t)row * N + col] = acc[t][i] + bias[row];
    }
  }
}

// ---------------------------------------------------------------------------
// kn2row implicit-GEMM 3x3 conv: y[Cout][B*H*W] = sum_{rc} Wp[rc] * shift(act)
// act NHWC bf16 [B][H][W][Cin]; Wp [rc][Cout][Cin] bf16. Cin%32==0.
// Pipelined over the Cin loop within each of the 9 taps.
// ---------------------------------------------------------------------------
__global__ __launch_bounds__(128) void conv_kn2row_wmma(
    float* __restrict__ y, const unsigned short* __restrict__ Wp,
    const unsigned short* __restrict__ act, const float* __restrict__ bias,
    int Cout, int Cin, int H, int W) {
  const int lane  = threadIdx.x & 31;
  const int wave  = threadIdx.x >> 5;
  const int m0    = (blockIdx.y * 4 + wave) * 16;
  const int n0    = blockIdx.x * 64;
  const int lm    = lane & 15;
  const int hiSel = lane >> 4;
  const int ko1   = hiSel * 8;
  const int HW    = H * W;
  const size_t N  = (size_t)128 * HW;

  int pb[4], ph[4], pw[4];
#pragma unroll
  for (int t = 0; t < 4; ++t) {
    int p = n0 + t * 16 + lm;
    pb[t] = p / HW;
    int rem = p - pb[t] * HW;
    ph[t] = rem / W;
    pw[t] = rem - ph[t] * W;
  }

  v8f z = {0.f, 0.f, 0.f, 0.f, 0.f, 0.f, 0.f, 0.f};
  v8f acc[4] = {z, z, z, z};

  for (int rc = 0; rc < 9; ++rc) {
    const int dr = rc / 3 - 1, dc = rc % 3 - 1;
    const unsigned short* arow =
        Wp + ((size_t)rc * Cout + (m0 + lm)) * Cin;
    if (rc < 8)   // warm L2/WGP$ for the next tap's weight row
      __builtin_prefetch(Wp + ((size_t)(rc + 1) * Cout + (m0 + lm)) * Cin, 0, 0);
    const unsigned short* brow[4];
    bool valid[4];
#pragma unroll
    for (int t = 0; t < 4; ++t) {
      int ih = ph[t] + dr, iw = pw[t] + dc;
      valid[t] = ((unsigned)ih < (unsigned)H) && ((unsigned)iw < (unsigned)W);
      brow[t] = valid[t]
                    ? act + ((size_t)(pb[t] * HW + ih * W + iw)) * Cin
                    : act;
    }
    // prologue for k = 0
    u32x8 aC = ldfrag(arow, ko1);
    u32x8 bC[4];
#pragma unroll
    for (int t = 0; t < 4; ++t) bC[t] = ldfrag_cond(brow[t], ko1, valid[t]);

    for (int k = 32; k < Cin; k += 32) {
      u32x8 aN = ldfrag(arow + k, ko1);
      u32x8 bN[4];
#pragma unroll
      for (int t = 0; t < 4; ++t)
        bN[t] = ldfrag_cond(brow[t] + k, ko1, valid[t]);
#pragma unroll
      for (int t = 0; t < 4; ++t) acc[t] = wmma_bf16(aC, bC[t], acc[t]);
      aC = aN;
#pragma unroll
      for (int t = 0; t < 4; ++t) bC[t] = bN[t];
    }
#pragma unroll
    for (int t = 0; t < 4; ++t) acc[t] = wmma_bf16(aC, bC[t], acc[t]);
  }

  const int rbase = m0 + hiSel * 8;
#pragma unroll
  for (int t = 0; t < 4; ++t) {
    const int col = n0 + t * 16 + lm;
#pragma unroll
    for (int i = 0; i < 8; ++i) {
      const int row = rbase + i;
      y[(size_t)row * N + col] = acc[t][i] + bias[row];
    }
  }
}

// ---------------------------------------------------------------------------
// BatchNorm: per-channel stats over y[ch][N] -> ss[2*ch]={scale,shift}
// ---------------------------------------------------------------------------
__global__ void bn_stats(const float* __restrict__ y,
                         const float* __restrict__ gamma,
                         const float* __restrict__ beta,
                         float* __restrict__ ss, int N) {
  const int ch = blockIdx.x;
  const float* row = y + (size_t)ch * N;
  float s = 0.f, sq = 0.f;
  for (int i = threadIdx.x; i < N; i += 256) {
    float v = row[i];
    s += v;
    sq += v * v;
  }
  __shared__ float sh[256], sh2[256];
  sh[threadIdx.x] = s;
  sh2[threadIdx.x] = sq;
  __syncthreads();
  for (int o = 128; o > 0; o >>= 1) {
    if (threadIdx.x < o) {
      sh[threadIdx.x] += sh[threadIdx.x + o];
      sh2[threadIdx.x] += sh2[threadIdx.x + o];
    }
    __syncthreads();
  }
  if (threadIdx.x == 0) {
    float mean = sh[0] / (float)N;
    float var  = sh2[0] / (float)N - mean * mean;
    float sc   = gamma[ch] * rsqrtf(var + 1e-5f);
    ss[2 * ch]     = sc;
    ss[2 * ch + 1] = beta[ch] - mean * sc;
  }
}

// normalize + ReLU with LDS-tiled transpose: y[C][Npix] -> act NHWC [Npix][C]
// tile = 32 channels x 64 pixels; coalesced reads and contiguous writes.
// grid = (Npix/64, C/32), block = 256
__global__ __launch_bounds__(256) void bn_relu_nhwc(
    const float* __restrict__ y, const float* __restrict__ ss,
    unsigned short* __restrict__ act, int Npix, int C) {
  __shared__ float tile[32][65];
  const int c0 = blockIdx.y * 32;
  const int p0 = blockIdx.x * 64;
  const int tx = threadIdx.x & 63;        // pixel within tile
  const int ty = threadIdx.x >> 6;        // 0..3
  for (int r = ty; r < 32; r += 4)
    tile[r][tx] = y[(size_t)(c0 + r) * Npix + (p0 + tx)];
  __syncthreads();
  const int cc = threadIdx.x & 31;
  const int pp = threadIdx.x >> 5;        // 0..7
  const float sc = ss[2 * (c0 + cc)];
  const float sh = ss[2 * (c0 + cc) + 1];
  for (int p = pp; p < 64; p += 8) {
    float v = tile[cc][p] * sc + sh;
    act[(size_t)(p0 + p) * C + (c0 + cc)] = f2bf(fmaxf(v, 0.f));
  }
}

// FC version: y[M][128] -> xb [128][M] bf16 (transposed for next GEMM's Bt)
__global__ void bn_relu_T(const float* __restrict__ y,
                          const float* __restrict__ ss,
                          unsigned short* __restrict__ xb, int logM) {
  int tid = blockIdx.x * blockDim.x + threadIdx.x;
  if (tid >= (128 << logM)) return;
  int m = tid & ((1 << logM) - 1);
  int b = tid >> logM;
  float v = y[(size_t)m * 128 + b] * ss[2 * m] + ss[2 * m + 1];
  xb[tid] = f2bf(fmaxf(v, 0.f));
}

// 2x2 max pool on NHWC bf16
__global__ void maxpool_nhwc(const unsigned short* __restrict__ src,
                             unsigned short* __restrict__ dst,
                             int Ho, int Wo, int logC) {
  int tid = blockIdx.x * blockDim.x + threadIdx.x;
  int total = (128 * Ho * Wo) << logC;
  if (tid >= total) return;
  int C  = 1 << logC;
  int c  = tid & (C - 1);
  int r  = tid >> logC;
  int wo = r % Wo; r /= Wo;
  int ho = r % Ho;
  int b  = r / Ho;
  int W = Wo * 2, H = Ho * 2;
  size_t base = ((((size_t)b * H + 2 * ho) * W) + 2 * wo) * C + c;
  size_t rs   = (size_t)W * C;
  float m = fmaxf(fmaxf(bf2f(src[base]), bf2f(src[base + C])),
                  fmaxf(bf2f(src[base + rs]), bf2f(src[base + rs + C])));
  dst[tid] = f2bf(m);
}

// NHWC [128][6][6][64] -> NCHW-flat [128][2304] bf16 (fc1 Bt)
__global__ void flatten_nchw(const unsigned short* __restrict__ src,
                             unsigned short* __restrict__ dst) {
  int tid = blockIdx.x * blockDim.x + threadIdx.x;   // 128*2304
  int c = tid & 63;
  int r = tid >> 6;
  int w = r % 6; r /= 6;
  int h = r % 6;
  int b = r / 6;
  dst[b * 2304 + c * 36 + h * 6 + w] = src[tid];
}

// ---------------------------------------------------------------------------
// Attention head + fc4 (tiny): one block per batch element
// o7: NHWC bf16 [128][12][12][64]; xb3: [128][128] bf16
// ---------------------------------------------------------------------------
__global__ __launch_bounds__(64) void attn_fc4(
    const unsigned short* __restrict__ o7,
    const unsigned short* __restrict__ xb3,
    const float* __restrict__ att_w, const float* __restrict__ att_b,
    const float* __restrict__ fc4_w, const float* __restrict__ fc4_b,
    float* __restrict__ out) {
  __shared__ float q[64], sc[144], g[64], red[2];
  const int b = blockIdx.x, t = threadIdx.x;

  { // q = o13 @ att_w.T + att_b
    const unsigned short* xr = xb3 + b * 128;
    float s = 0.f;
    for (int i = 0; i < 128; ++i) s += att_w[t * 128 + i] * bf2f(xr[i]);
    q[t] = s + att_b[t];
  }
  __syncthreads();
  const unsigned short* ob = o7 + (size_t)b * 144 * 64;
  for (int p = t; p < 144; p += 64) {
    const unsigned short* pr = ob + p * 64;
    float s = 0.f;
    for (int c = 0; c < 64; ++c) s += bf2f(pr[c]) * q[c];
    sc[p] = s;
  }
  __syncthreads();
  if (t == 0) {
    float m = sc[0];
    for (int p = 1; p < 144; ++p) m = fmaxf(m, sc[p]);
    red[0] = m;
  }
  __syncthreads();
  for (int p = t; p < 144; p += 64) sc[p] = __expf(sc[p] - red[0]);
  __syncthreads();
  if (t == 0) {
    float s = 0.f;
    for (int p = 0; p < 144; ++p) s += sc[p];
    red[1] = 1.f / s;
  }
  __syncthreads();
  {
    float gc = 0.f;
    for (int p = 0; p < 144; ++p) gc += sc[p] * bf2f(ob[p * 64 + t]);
    g[t] = gc * red[1];
  }
  __syncthreads();
  if (t < 7) {
    float o = fc4_b[t];
    for (int c = 0; c < 64; ++c) o += fc4_w[t * 64 + c] * g[c];
    out[b * 7 + t] = o;
  }
}

// ---------------------------------------------------------------------------
// Host-side orchestration
// ---------------------------------------------------------------------------
extern "C" void kernel_launch(void* const* d_in, const int* in_sizes, int n_in,
                              void* d_out, int out_size, void* d_ws,
                              size_t ws_size, hipStream_t stream) {
  (void)in_sizes; (void)n_in; (void)out_size; (void)ws_size;

  const float* X = (const float*)d_in[0];
  const float *Wc[7], *Bc[7], *Gc[7], *Btc[7];
  for (int i = 1; i <= 6; ++i) {
    int base = 1 + (i - 1) * 4;
    Wc[i]  = (const float*)d_in[base + 0];
    Bc[i]  = (const float*)d_in[base + 1];
    Gc[i]  = (const float*)d_in[base + 2];
    Btc[i] = (const float*)d_in[base + 3];
  }
  const float* fc1w = (const float*)d_in[25];
  const float* fc1b = (const float*)d_in[26];
  const float* fc1g = (const float*)d_in[27];
  const float* fc1t = (const float*)d_in[28];
  const float* fc2w = (const float*)d_in[29];
  const float* fc2b = (const float*)d_in[30];
  const float* fc2g = (const float*)d_in[31];
  const float* fc2t = (const float*)d_in[32];
  const float* fc3w = (const float*)d_in[33];
  const float* fc3b = (const float*)d_in[34];
  const float* fc3g = (const float*)d_in[35];
  const float* fc3t = (const float*)d_in[36];
  const float* attw = (const float*)d_in[37];
  const float* attb = (const float*)d_in[38];
  const float* fc4w = (const float*)d_in[39];
  const float* fc4b = (const float*)d_in[40];

  const int N1 = 128 * 48 * 48;   // 294912
  const int N2 = 128 * 24 * 24;   // 73728
  const int N3 = 128 * 12 * 12;   // 18432

  // workspace carve-up
  char* base = (char*)d_ws;
  size_t off = 0;
  auto alloc = [&](size_t bytes) {
    void* p = base + off;
    off = (off + bytes + 255) & ~(size_t)255;
    return p;
  };
  float*          y     = (float*)alloc((size_t)256 * N1 * 4);
  unsigned short* actA  = (unsigned short*)alloc((size_t)N1 * 256 * 2);
  unsigned short* actB  = (unsigned short*)alloc((size_t)N1 * 256 * 2);
  unsigned short* col1  = (unsigned short*)alloc((size_t)N1 * 32 * 2);
  unsigned short* W1p   = (unsigned short*)alloc(256 * 32 * 2);
  unsigned short* Wp2   = (unsigned short*)alloc((size_t)589824 * 2);
  unsigned short* Wp3   = (unsigned short*)alloc((size_t)294912 * 2);
  unsigned short* Wp4   = (unsigned short*)alloc((size_t)147456 * 2);
  unsigned short* Wp5   = (unsigned short*)alloc((size_t)73728 * 2);
  unsigned short* Wp6   = (unsigned short*)alloc((size_t)36864 * 2);
  unsigned short* f1wb  = (unsigned short*)alloc((size_t)1179648 * 2);
  unsigned short* f2wb  = (unsigned short*)alloc((size_t)131072 * 2);
  unsigned short* f3wb  = (unsigned short*)alloc((size_t)32768 * 2);
  unsigned short* pool3 = (unsigned short*)alloc((size_t)128 * 36 * 64 * 2);
  unsigned short* xflat = (unsigned short*)alloc((size_t)128 * 2304 * 2);
  unsigned short* xb1   = (unsigned short*)alloc((size_t)128 * 512 * 2);
  unsigned short* xb2   = (unsigned short*)alloc((size_t)128 * 256 * 2);
  unsigned short* xb3   = (unsigned short*)alloc((size_t)128 * 128 * 2);
  float*          ss    = (float*)alloc(2 * 512 * 4);

  const int T = 256;
  // ---- weight packing ----
  pack_w1<<<32, T, 0, stream>>>(W1p, Wc[1]);
  pack_conv_w<<<589824 / T, T, 0, stream>>>(Wp2, Wc[2], 256, 256, 589824);
  pack_conv_w<<<294912 / T, T, 0, stream>>>(Wp3, Wc[3], 128, 256, 294912);
  pack_conv_w<<<147456 / T, T, 0, stream>>>(Wp4, Wc[4], 128, 128, 147456);
  pack_conv_w<<< 73728 / T, T, 0, stream>>>(Wp5, Wc[5],  64, 128,  73728);
  pack_conv_w<<< 36864 / T, T, 0, stream>>>(Wp6, Wc[6],  64,  64,  36864);
  pack_bf16<<<1179648 / T, T, 0, stream>>>(f1wb, fc1w, 1179648);
  pack_bf16<<< 131072 / T, T, 0, stream>>>(f2wb, fc2w, 131072);
  pack_bf16<<<  32768 / T, T, 0, stream>>>(f3wb, fc3w, 32768);

  // ---- conv1 (im2col + WMMA GEMM, K padded to 32) ----
  im2col_conv1<<<(N1 * 32) / T, T, 0, stream>>>(col1, X);
  gemm_wmma<<<dim3(N1 / 64, 4), 128, 0, stream>>>(y, W1p, col1, Bc[1], N1, 32);
  bn_stats<<<256, T, 0, stream>>>(y, Gc[1], Btc[1], ss, N1);
  bn_relu_nhwc<<<dim3(N1 / 64, 8), T, 0, stream>>>(y, ss, actA, N1, 256);

  // ---- conv2 ----
  conv_kn2row_wmma<<<dim3(N1 / 64, 4), 128, 0, stream>>>(y, Wp2, actA, Bc[2],
                                                         256, 256, 48, 48);
  bn_stats<<<256, T, 0, stream>>>(y, Gc[2], Btc[2], ss, N1);
  bn_relu_nhwc<<<dim3(N1 / 64, 8), T, 0, stream>>>(y, ss, actB, N1, 256);
  maxpool_nhwc<<<(N2 * 256) / T, T, 0, stream>>>(actB, actA, 24, 24, 8);

  // ---- conv3 / conv4 ----
  conv_kn2row_wmma<<<dim3(N2 / 64, 2), 128, 0, stream>>>(y, Wp3, actA, Bc[3],
                                                         128, 256, 24, 24);
  bn_stats<<<128, T, 0, stream>>>(y, Gc[3], Btc[3], ss, N2);
  bn_relu_nhwc<<<dim3(N2 / 64, 4), T, 0, stream>>>(y, ss, actB, N2, 128);
  conv_kn2row_wmma<<<dim3(N2 / 64, 2), 128, 0, stream>>>(y, Wp4, actB, Bc[4],
                                                         128, 128, 24, 24);
  bn_stats<<<128, T, 0, stream>>>(y, Gc[4], Btc[4], ss, N2);
  bn_relu_nhwc<<<dim3(N2 / 64, 4), T, 0, stream>>>(y, ss, actA, N2, 128);
  maxpool_nhwc<<<(N3 * 128) / T, T, 0, stream>>>(actA, actB, 12, 12, 7);

  // ---- conv5 / conv6 ----
  conv_kn2row_wmma<<<dim3(N3 / 64, 1), 128, 0, stream>>>(y, Wp5, actB, Bc[5],
                                                         64, 128, 12, 12);
  bn_stats<<<64, T, 0, stream>>>(y, Gc[5], Btc[5], ss, N3);
  bn_relu_nhwc<<<dim3(N3 / 64, 2), T, 0, stream>>>(y, ss, actA, N3, 64); // o7
  conv_kn2row_wmma<<<dim3(N3 / 64, 1), 128, 0, stream>>>(y, Wp6, actA, Bc[6],
                                                         64, 64, 12, 12);
  bn_stats<<<64, T, 0, stream>>>(y, Gc[6], Btc[6], ss, N3);
  bn_relu_nhwc<<<dim3(N3 / 64, 2), T, 0, stream>>>(y, ss, actB, N3, 64);
  maxpool_nhwc<<<(128 * 36 * 64) / T, T, 0, stream>>>(actB, pool3, 6, 6, 6);
  flatten_nchw<<<(128 * 2304) / T, T, 0, stream>>>(pool3, xflat);

  // ---- FC stack (WMMA GEMMs, N = batch = 128) ----
  gemm_wmma<<<dim3(2, 8), 128, 0, stream>>>(y, f1wb, xflat, fc1b, 128, 2304);
  bn_stats<<<512, T, 0, stream>>>(y, fc1g, fc1t, ss, 128);
  bn_relu_T<<<(128 * 512) / T, T, 0, stream>>>(y, ss, xb1, 9);

  gemm_wmma<<<dim3(2, 4), 128, 0, stream>>>(y, f2wb, xb1, fc2b, 128, 512);
  bn_stats<<<256, T, 0, stream>>>(y, fc2g, fc2t, ss, 128);
  bn_relu_T<<<(128 * 256) / T, T, 0, stream>>>(y, ss, xb2, 8);

  gemm_wmma<<<dim3(2, 2), 128, 0, stream>>>(y, f3wb, xb2, fc3b, 128, 256);
  bn_stats<<<128, T, 0, stream>>>(y, fc3g, fc3t, ss, 128);
  bn_relu_T<<<(128 * 128) / T, T, 0, stream>>>(y, ss, xb3, 7);

  // ---- attention + fc4 ----
  attn_fc4<<<128, 64, 0, stream>>>(actA, xb3, attw, attb, fc4w, fc4b,
                                   (float*)d_out);
}
